// continuous_seg_nonspeed_171798692346
// MI455X (gfx1250) — matrix operand order repaired and checked by reading, not compiled, not run on hardware
//
#include <hip/hip_runtime.h>
#include <hip/hip_bf16.h>
#include <stdint.h>

// ---------------- CDNA5 WMMA types ----------------
typedef __attribute__((ext_vector_type(16))) __bf16 v16bf;
typedef __attribute__((ext_vector_type(8)))  float  v8f;

#define B_     64
#define T_     64
#define S_     128
#define H_     128
#define NG     512      // 4*H gates
#define KENC   416      // 3*128 + x(3) + bias(1) + pad -> mult of 32
#define KDEC   544      // 4*128 + bias(1) + pad
#define KEMB   160      // 128 + bias(1) + pad
#define ROWS   8192     // B*S
#define TILE_M 64
#define TPRED  4
#define TOUT   59       // T - TPRED - 1

__device__ __forceinline__ float sigm(float x) { return 1.0f / (1.0f + __expf(-x)); }

// A fragment: 16x32 bf16, A stored row-major [M][Kpad] in LDS.
// Lane L: g=L>>4, m=L&15. elems 0..7 = K 8g..8g+7 ; elems 8..15 = K 16+8g..16+8g+7.
__device__ __forceinline__ v16bf ldsA(const __bf16* base, int kpad, int mbase, int kc, int lane) {
  int g = lane >> 4, m = lane & 15;
  const __bf16* p = base + (mbase + m) * kpad + kc + 8 * g;
  v16bf a;
  reinterpret_cast<uint4*>(&a)[0] = *reinterpret_cast<const uint4*>(p);
  reinterpret_cast<uint4*>(&a)[1] = *reinterpret_cast<const uint4*>(p + 16);
  return a;
}

// B fragment: 32x16 bf16, weights stored transposed [N][Kpad] in global.
// Lane L: g=L>>4, n=L&15. elems 0..7 = K 8g..8g+7 ; elems 8..15 = K 16+8g..16+8g+7, col n.
__device__ __forceinline__ v16bf ldgB(const __bf16* WT, int kpad, int n0, int kc, int lane) {
  int g = lane >> 4, n = lane & 15;
  const __bf16* p = WT + (size_t)(n0 + n) * kpad + kc + 8 * g;
  v16bf b;
  reinterpret_cast<uint4*>(&b)[0] = reinterpret_cast<const uint4*>(p)[0];
  reinterpret_cast<uint4*>(&b)[1] = reinterpret_cast<const uint4*>(p + 16)[0];
  return b;
}

__device__ __forceinline__ v8f wmma_bf16(v16bf a, v16bf b, v8f c) {
  return __builtin_amdgcn_wmma_f32_16x16x32_bf16(false, a, false, b, (short)0, c, false, false);
}

// ---------------- weight packing ----------------
extern "C" __global__ void pack_encw(const float* __restrict__ Wx, const float* __restrict__ Wh,
                                     const float* __restrict__ Wa, const float* __restrict__ Wb,
                                     const float* __restrict__ b, __bf16* __restrict__ out) {
  int i = blockIdx.x * blockDim.x + threadIdx.x;
  if (i >= NG * KENC) return;
  int n = i / KENC, k = i - n * KENC;
  float v = 0.f;
  if      (k < 128) v = Wh[k * NG + n];
  else if (k < 256) v = Wa[(k - 128) * NG + n];
  else if (k < 384) v = Wb[(k - 256) * NG + n];
  else if (k < 387) v = Wx[(k - 384) * NG + n];
  else if (k == 387) v = b[n];
  out[i] = (__bf16)v;
}

extern "C" __global__ void pack_decw(const float* __restrict__ Wx, const float* __restrict__ Wh,
                                     const float* __restrict__ Wa, const float* __restrict__ Wb,
                                     const float* __restrict__ b, __bf16* __restrict__ out) {
  int i = blockIdx.x * blockDim.x + threadIdx.x;
  if (i >= NG * KDEC) return;
  int n = i / KDEC, k = i - n * KDEC;
  float v = 0.f;
  if      (k < 128) v = Wx[k * NG + n];
  else if (k < 256) v = Wh[(k - 128) * NG + n];
  else if (k < 384) v = Wa[(k - 256) * NG + n];
  else if (k < 512) v = Wb[(k - 384) * NG + n];
  else if (k == 512) v = b[n];
  out[i] = (__bf16)v;
}

extern "C" __global__ void pack_embw(const float* __restrict__ W, const float* __restrict__ b,
                                     __bf16* __restrict__ out) {
  int i = blockIdx.x * blockDim.x + threadIdx.x;
  if (i >= H_ * KEMB) return;
  int n = i / KEMB, k = i - n * KEMB;
  float v = 0.f;
  if (k < 128) v = W[k * H_ + n];
  else if (k == 128) v = b[n];
  out[i] = (__bf16)v;
}

extern "C" __global__ void zero_ws(uint32_t* __restrict__ p, int n) {
  int i = blockIdx.x * blockDim.x + threadIdx.x;
  if (i < n) p[i] = 0u;
}

// ---------------- encoder step + embedding (fused) ----------------
extern "C" __global__ void __launch_bounds__(512)
enc_step(const float* __restrict__ in, int t,
         const __bf16* __restrict__ WencT, const __bf16* __restrict__ WembT,
         const __bf16* __restrict__ eh_in, __bf16* __restrict__ eh_out,
         float* __restrict__ ec, __bf16* __restrict__ code)
{
  extern __shared__ char smem[];
  __bf16* Alds = reinterpret_cast<__bf16*>(smem);     // [64][KENC]; later aliased as A2 [64][KEMB]
  const int tid  = threadIdx.x;
  const int lane = tid & 31;
  const int w    = tid >> 5;        // 0..15
  const int wh   = w & 7;           // h-subtile: cols 16*wh (per gate)
  const int mh   = w >> 3;          // m half: rows 32*mh..32*mh+31
  const int m0t  = blockIdx.x * TILE_M;

  // stage A = [eh | eh_after | eh_before] via 16B chunks (64 rows x 3 segs x 8 chunks)
  #pragma unroll
  for (int j = 0; j < 3; ++j) {
    int c = tid + 512 * j;          // 0..1535
    int part = c & 7;
    int segr = c >> 3;              // 0..191
    int m = segr / 3;
    int shift = segr - 3 * m;
    int rg = m0t + m;
    int s = rg & (S_ - 1);
    uint4 v = make_uint4(0u, 0u, 0u, 0u);
    int src = rg;
    bool ok = true;
    if (shift == 1)      { ok = (s < S_ - 1); src = rg + 1; }
    else if (shift == 2) { ok = (s > 0);      src = rg - 1; }
    if (ok) v = reinterpret_cast<const uint4*>(eh_in + (size_t)src * H_)[part];
    reinterpret_cast<uint4*>(Alds + m * KENC + shift * H_)[part] = v;
  }
  if (tid < TILE_M) {               // x features + bias-one + pad
    int m = tid, rg = m0t + m, b = rg >> 7, s = rg & (S_ - 1);
    const float* xp = in + ((size_t)(b * T_ + t) * S_ + s) * 3;
    __bf16* dst = Alds + m * KENC + 384;
    dst[0] = (__bf16)xp[0]; dst[1] = (__bf16)xp[1]; dst[2] = (__bf16)xp[2];
    dst[3] = (__bf16)1.0f;
    for (int k = 4; k < 32; ++k) dst[k] = (__bf16)0.0f;
  }
  __syncthreads();

  // gates GEMM: [64 x 416] @ [416 x 512], each wave owns cols {h0, h0+128, h0+256, h0+384}
  v8f zero = {0.f,0.f,0.f,0.f,0.f,0.f,0.f,0.f};
  v8f acc[2][4];
  #pragma unroll
  for (int msi = 0; msi < 2; ++msi)
    #pragma unroll
    for (int q = 0; q < 4; ++q) acc[msi][q] = zero;

  for (int kc = 0; kc < KENC; kc += 32) {
    v16bf a0 = ldsA(Alds, KENC, 32 * mh,      kc, lane);
    v16bf a1 = ldsA(Alds, KENC, 32 * mh + 16, kc, lane);
    __builtin_prefetch(WencT + (size_t)(16 * wh + (lane & 15)) * KENC + kc + 32, 0, 1);
    #pragma unroll
    for (int q = 0; q < 4; ++q) {
      v16bf bq = ldgB(WencT, KENC, 128 * q + 16 * wh, kc, lane);
      acc[0][q] = wmma_bf16(a0, bq, acc[0][q]);
      acc[1][q] = wmma_bf16(a1, bq, acc[1][q]);
    }
  }
  __syncthreads();                  // Alds dead; reuse LDS as A2

  __bf16* A2 = reinterpret_cast<__bf16*>(smem);       // [64][KEMB]
  const int g2 = lane >> 4, nl = lane & 15;
  const int h = 16 * wh + nl;
  #pragma unroll
  for (int msi = 0; msi < 2; ++msi) {
    int rbase = 32 * mh + 16 * msi + 8 * g2;
    #pragma unroll
    for (int r = 0; r < 8; ++r) {
      int rl = rbase + r;
      size_t idx = (size_t)(m0t + rl) * H_ + h;
      float iv = acc[msi][0][r], fv = acc[msi][1][r];
      float gv = acc[msi][2][r], ov = acc[msi][3][r];
      float cn = sigm(fv) * ec[idx] + sigm(iv) * tanhf(gv);
      float hn = sigm(ov) * tanhf(cn);
      ec[idx]     = cn;
      eh_out[idx] = (__bf16)hn;
      A2[rl * KEMB + h] = (__bf16)hn;
    }
  }
  if (tid < TILE_M) {               // bias-one + pad cols for embedding GEMM
    __bf16* dst = A2 + tid * KEMB + H_;
    dst[0] = (__bf16)1.0f;
    for (int k = 1; k < 32; ++k) dst[k] = (__bf16)0.0f;
  }
  __syncthreads();

  // embedding GEMM: code = sigmoid([64 x 160] @ [160 x 128])
  v8f a2c[2]; a2c[0] = zero; a2c[1] = zero;
  for (int kc = 0; kc < KEMB; kc += 32) {
    v16bf a0 = ldsA(A2, KEMB, 32 * mh,      kc, lane);
    v16bf a1 = ldsA(A2, KEMB, 32 * mh + 16, kc, lane);
    v16bf bq = ldgB(WembT, KEMB, 16 * wh, kc, lane);
    a2c[0] = wmma_bf16(a0, bq, a2c[0]);
    a2c[1] = wmma_bf16(a1, bq, a2c[1]);
  }
  #pragma unroll
  for (int msi = 0; msi < 2; ++msi) {
    int rbase = 32 * mh + 16 * msi + 8 * g2;
    #pragma unroll
    for (int r = 0; r < 8; ++r) {
      size_t idx = (size_t)(m0t + rbase + r) * H_ + h;
      code[idx] = (__bf16)sigm(a2c[msi][r]);
    }
  }
}

// ---------------- decoder step + output head (fused) ----------------
extern "C" __global__ void __launch_bounds__(512)
dec_step(const __bf16* __restrict__ WdecT,
         const __bf16* __restrict__ code,
         const __bf16* __restrict__ dh_in, __bf16* __restrict__ dh_out,
         float* __restrict__ dh32,
         const float* __restrict__ outW, const float* __restrict__ outB,
         float* __restrict__ outbuf)
{
  extern __shared__ char smem[];
  __bf16* Alds = reinterpret_cast<__bf16*>(smem);     // [64][KDEC]; later aliased as dh_lds f32
  const int tid = threadIdx.x, lane = tid & 31, w = tid >> 5;
  const int wh = w & 7, mh = w >> 3;
  const int m0t = blockIdx.x * TILE_M;

  // stage A = [code | dh | dh_after | dh_before] (64 rows x 4 segs x 8 chunks)
  #pragma unroll
  for (int j = 0; j < 4; ++j) {
    int c = tid + 512 * j;          // 0..2047
    int part = c & 7;
    int segr = c >> 3;              // 0..255
    int m = segr >> 2, seg = segr & 3;
    int rg = m0t + m, s = rg & (S_ - 1);
    uint4 v = make_uint4(0u, 0u, 0u, 0u);
    if (seg == 0) {
      v = reinterpret_cast<const uint4*>(code + (size_t)rg * H_)[part];
    } else {
      int src = rg; bool ok = true;
      if (seg == 2)      { ok = (s < S_ - 1); src = rg + 1; }
      else if (seg == 3) { ok = (s > 0);      src = rg - 1; }
      if (ok) v = reinterpret_cast<const uint4*>(dh_in + (size_t)src * H_)[part];
    }
    reinterpret_cast<uint4*>(Alds + m * KDEC + seg * H_)[part] = v;
  }
  if (tid < TILE_M) {
    __bf16* dst = Alds + tid * KDEC + 512;
    dst[0] = (__bf16)1.0f;
    for (int k = 1; k < 32; ++k) dst[k] = (__bf16)0.0f;
  }
  __syncthreads();

  v8f zero = {0.f,0.f,0.f,0.f,0.f,0.f,0.f,0.f};
  v8f acc[2][4];
  #pragma unroll
  for (int msi = 0; msi < 2; ++msi)
    #pragma unroll
    for (int q = 0; q < 4; ++q) acc[msi][q] = zero;

  for (int kc = 0; kc < KDEC; kc += 32) {
    v16bf a0 = ldsA(Alds, KDEC, 32 * mh,      kc, lane);
    v16bf a1 = ldsA(Alds, KDEC, 32 * mh + 16, kc, lane);
    __builtin_prefetch(WdecT + (size_t)(16 * wh + (lane & 15)) * KDEC + kc + 32, 0, 1);
    #pragma unroll
    for (int q = 0; q < 4; ++q) {
      v16bf bq = ldgB(WdecT, KDEC, 128 * q + 16 * wh, kc, lane);
      acc[0][q] = wmma_bf16(a0, bq, acc[0][q]);
      acc[1][q] = wmma_bf16(a1, bq, acc[1][q]);
    }
  }
  __syncthreads();                  // Alds dead; reuse as dh_lds

  float* dhl = reinterpret_cast<float*>(smem);        // [64][128]
  const int g2 = lane >> 4, nl = lane & 15;
  const int h = 16 * wh + nl;
  #pragma unroll
  for (int msi = 0; msi < 2; ++msi) {
    int rbase = 32 * mh + 16 * msi + 8 * g2;
    #pragma unroll
    for (int r = 0; r < 8; ++r) {
      int rl = rbase + r;
      size_t idx = (size_t)(m0t + rl) * H_ + h;
      float iv = acc[msi][0][r], fv = acc[msi][1][r];
      float gv = acc[msi][2][r], ov = acc[msi][3][r];
      float c_old = dh32[idx];      // faithful to source bug: cell state = dh
      float cn = sigm(fv) * c_old + sigm(iv) * tanhf(gv);
      float hn = sigm(ov) * tanhf(cn);
      dh32[idx]   = hn;
      dh_out[idx] = (__bf16)hn;
      dhl[rl * H_ + h] = hn;
    }
  }
  __syncthreads();

  if (tid < TILE_M) {               // out = dh @ out_W + out_b
    float a = outB[0];
    const float* row = dhl + tid * H_;
    #pragma unroll 4
    for (int k = 0; k < H_; ++k) a += row[k] * outW[k];
    outbuf[m0t + tid] = a;
  }
}

// ---------------- next_data assembly / output write ----------------
extern "C" __global__ void assemble_step(const float* __restrict__ in, int t,
                                         const float* __restrict__ outbuf,
                                         float* __restrict__ out)
{
  int i = blockIdx.x * blockDim.x + threadIdx.x;
  if (i >= ROWS) return;
  int b = i >> 7, s = i & (S_ - 1);
  float o = outbuf[i];
  float In = (s == 0) ? in[((size_t)(b * T_ + (t + 1)) * S_ + 0) * 3 + 1]
                      : outbuf[i - 1];
  float num = in[((size_t)(b * T_ + t) * S_ + s) * 3 + 2] + In - o;
  if (t >= TPRED) {
    size_t base = ((size_t)(b * TOUT + (t - TPRED)) * S_ + s) * 3;
    out[base + 0] = o;
    out[base + 1] = In;
    out[base + 2] = num;
  }
}

// ---------------- host orchestration ----------------
extern "C" void kernel_launch(void* const* d_in, const int* in_sizes, int n_in,
                              void* d_out, int out_size, void* d_ws, size_t ws_size,
                              hipStream_t stream)
{
  (void)in_sizes; (void)n_in; (void)out_size; (void)ws_size;
  const float* in   = (const float*)d_in[0];
  const float* eWx  = (const float*)d_in[1];
  const float* eWh  = (const float*)d_in[2];
  const float* eWa  = (const float*)d_in[3];
  const float* eWb  = (const float*)d_in[4];
  const float* eb   = (const float*)d_in[5];
  const float* dWx  = (const float*)d_in[6];
  const float* dWh  = (const float*)d_in[7];
  const float* dWa  = (const float*)d_in[8];
  const float* dWb  = (const float*)d_in[9];
  const float* db   = (const float*)d_in[10];
  const float* embW = (const float*)d_in[11];
  const float* embB = (const float*)d_in[12];
  const float* outW = (const float*)d_in[13];
  const float* outB = (const float*)d_in[14];

  char* ws = (char*)d_ws;
  __bf16* WencT = (__bf16*)(ws + 0);         // 512*416*2 = 425984
  __bf16* WdecT = (__bf16*)(ws + 425984);    // 512*544*2 = 557056
  __bf16* WembT = (__bf16*)(ws + 983040);    // 128*160*2 = 40960
  __bf16* codeB = (__bf16*)(ws + 1024000);   // 8192*128*2 = 2097152
  float*  outbf = (float*) (ws + 3121152);   // 8192*4     = 32768
  __bf16* eh0   = (__bf16*)(ws + 3153920);   // 2097152  (zeroed)
  __bf16* dh0   = (__bf16*)(ws + 5251072);   // 2097152  (zeroed)
  float*  dh32  = (float*) (ws + 7348224);   // 4194304  (zeroed)
  float*  ec    = (float*) (ws + 11542528);  // 4194304  (zeroed)
  __bf16* eh1   = (__bf16*)(ws + 15736832);  // 2097152
  __bf16* dh1   = (__bf16*)(ws + 17833984);  // 2097152  -> total ~19.0 MiB

  __bf16* eh[2]  = { eh0, eh1 };
  __bf16* dhb[2] = { dh0, dh1 };

  hipLaunchKernelGGL(pack_encw, dim3((NG * KENC + 255) / 256), dim3(256), 0, stream,
                     eWx, eWh, eWa, eWb, eb, WencT);
  hipLaunchKernelGGL(pack_decw, dim3((NG * KDEC + 255) / 256), dim3(256), 0, stream,
                     dWx, dWh, dWa, dWb, db, WdecT);
  hipLaunchKernelGGL(pack_embw, dim3((H_ * KEMB + 255) / 256), dim3(256), 0, stream,
                     embW, embB, WembT);
  hipLaunchKernelGGL(zero_ws, dim3(3145728 / 256), dim3(256), 0, stream,
                     (uint32_t*)(ws + 3153920), 3145728);   // eh0|dh0|dh32|ec contiguous

  int cur = 0;
  for (int t = 0; t < T_ - 1; ++t) {
    hipLaunchKernelGGL(enc_step, dim3(ROWS / TILE_M), dim3(512), TILE_M * KENC * 2, stream,
                       in, t, WencT, WembT, eh[cur], eh[1 - cur], ec, codeB);
    hipLaunchKernelGGL(dec_step, dim3(ROWS / TILE_M), dim3(512), TILE_M * KDEC * 2, stream,
                       WdecT, codeB, dhb[cur], dhb[1 - cur], dh32, outW, outB, outbf);
    hipLaunchKernelGGL(assemble_step, dim3(ROWS / 256), dim3(256), 0, stream,
                       in, t, outbf, (float*)d_out);
    cur ^= 1;
  }
}
